// PointTransformerEncoder_36369783063229
// MI455X (gfx1250) — compile-verified, matrix operand order checked by reading, not run on hardware
//
#include <hip/hip_runtime.h>
#include <math.h>

// ---------------------------------------------------------------------------
// PointTransformer encoder for MI455X (gfx1250, wave32).
//
// Dense linears: f32 master tensors, pre-packed into WMMA-fragment-layout f16
// tile buffers, then branch-free WMMA GEMMs (v_wmma_f32_16x16x32_f16, f32
// accumulate). Big GEMMs (N%64==0) use 8-wave workgroups computing a 128x64
// tile with B fragments staged through double-buffered LDS (ds_load_b128
// reuse across all 8 waves); odd shapes use a 1-wave 16x16 path. All
// activations are L2-resident (192MB); HBM is touched once per tensor.
// ---------------------------------------------------------------------------

#define EPSV 1e-5f

typedef __attribute__((ext_vector_type(16))) _Float16 v16h;
typedef __attribute__((ext_vector_type(8)))  float    v8f;

static inline int cdiv(long long a, long long b) { return (int)((a + b - 1) / b); }

// ---------------------------------------------------------------------------
// Fragment packing.
// A 16x32 f16 WMMA fragment: lane L holds row M=L%16; kb=(L/16)*8;
// element e<8 -> K=kb+e, e>=8 -> K=16+kb+(e-8). One lane's 16 halves are two
// 8-float contiguous runs of the source row; packed so the GEMM reads one
// contiguous v16h (32B) per lane per tile: addr = ((mt*KT+kt)*32 + lane).
// ---------------------------------------------------------------------------
__global__ __launch_bounds__(256) void pack_a_kernel(
    const float* __restrict__ A, int lda, int M, int K, int KT,
    v16h* __restrict__ Ap, int total)
{
  int t = blockIdx.x * blockDim.x + threadIdx.x;
  if (t >= total) return;
  int lane = t & 31;
  int tile = t >> 5;
  int kt = tile % KT;
  int mt = tile / KT;
  int row = mt * 16 + (lane & 15);
  int kb  = (lane >> 4) << 3;
  v16h v;
  if (row < M && (kt * 32 + 31) < K) {
    const float* s0 = A + (size_t)row * lda + kt * 32 + kb;
#pragma unroll
    for (int e = 0; e < 16; ++e)
      v[e] = (_Float16)((e < 8) ? s0[e] : s0[8 + e]);
  } else {
#pragma unroll
    for (int e = 0; e < 16; ++e) {
      int k = kt * 32 + kb + ((e < 8) ? e : (8 + e));
      v[e] = (_Float16)((row < M && k < K) ? A[(size_t)row * lda + k] : 0.0f);
    }
  }
  Ap[t] = v;
}

// B fragment mirrors A with N across lanes: lane L holds col N=L%16; same K
// pattern. Source W is row-major (K x N): strided reads, one-time pack cost.
__global__ __launch_bounds__(256) void pack_b_kernel(
    const float* __restrict__ W, int ldw, int N, int K, int KT,
    v16h* __restrict__ Bp, int total)
{
  int t = blockIdx.x * blockDim.x + threadIdx.x;
  if (t >= total) return;
  int lane = t & 31;
  int tile = t >> 5;
  int kt = tile % KT;
  int nt = tile / KT;
  int col = nt * 16 + (lane & 15);
  int kb  = (lane >> 4) << 3;
  v16h v;
  if (col < N && (kt * 32 + 31) < K) {
    const float* s0 = W + (size_t)(kt * 32 + kb) * ldw + col;
#pragma unroll
    for (int e = 0; e < 16; ++e) {
      int ko = (e < 8) ? e : (8 + e);
      v[e] = (_Float16)(s0[(size_t)ko * ldw]);
    }
  } else {
#pragma unroll
    for (int e = 0; e < 16; ++e) {
      int k = kt * 32 + kb + ((e < 8) ? e : (8 + e));
      v[e] = (_Float16)((col < N && k < K) ? W[(size_t)k * ldw + col] : 0.0f);
    }
  }
  Bp[t] = v;
}

// ---------------------------------------------------------------------------
// 8-wave LDS-tiled WMMA GEMM (N%64==0): block computes 128x64. B tiles (4KB
// per K-step) are staged into double-buffered LDS by the first 4 waves and
// re-read by all 8 waves (8x B-traffic reduction); A streams per-wave from
// the packed global buffer with prefetch. EXEC is all-ones at every wmma.
// ---------------------------------------------------------------------------
template <int TN>
__global__ __launch_bounds__(256) void gemm_wmma_lds(
    const v16h* __restrict__ Ap, const v16h* __restrict__ Bp,
    const float* __restrict__ bias, float* __restrict__ C, int ldc,
    int M, int N, int KT, int MT)
{
  __shared__ v16h sB[2][TN * 32];
  const int tid  = threadIdx.x;
  const int lane = tid & 31;
  const int wave = tid >> 5;
  const int mt   = blockIdx.y * 8 + wave;
  const int mtc  = (mt < MT) ? mt : (MT - 1);  // clamp loads; stores row-guarded
  const int nt0  = blockIdx.x * TN;
  const v16h* ap = Ap + (size_t)mtc * KT * 32 + lane;

  v8f acc[TN];
#pragma unroll
  for (int t = 0; t < TN; ++t) acc[t] = (v8f){};

  if (tid < TN * 32)
    sB[0][tid] = Bp[((size_t)(nt0 + (tid >> 5)) * KT) * 32 + (tid & 31)];
  __syncthreads();

  int buf = 0;
  for (int kt = 0; kt < KT; ++kt) {
    if (kt + 1 < KT) {
      if (tid < TN * 32)
        sB[buf ^ 1][tid] =
            Bp[((size_t)(nt0 + (tid >> 5)) * KT + (kt + 1)) * 32 + (tid & 31)];
      __builtin_prefetch(ap + (size_t)(kt + 1) * 32, 0, 1);  // global_prefetch_b8
    }
    v16h a = ap[(size_t)kt * 32];
#pragma unroll
    for (int t = 0; t < TN; ++t) {
      v16h b = sB[buf][t * 32 + lane];
      acc[t] = __builtin_amdgcn_wmma_f32_16x16x32_f16(
          false, a, false, b, (short)0, acc[t], false, false);
    }
    __syncthreads();
    buf ^= 1;
  }

  const int l16 = lane & 15;
  const int rbase = mt * 16 + ((lane >> 4) << 3);  // D: vgpr r -> M=rbase+r
#pragma unroll
  for (int t = 0; t < TN; ++t) {
    const int col = (nt0 + t) * 16 + l16;
    if (col < N) {
      const float bv = bias ? bias[col] : 0.0f;
#pragma unroll
      for (int r = 0; r < 8; ++r) {
        const int rr = rbase + r;
        if (rr < M) C[(size_t)rr * ldc + col] = acc[t][r] + bv;
      }
    }
  }
}

// 1-wave packed WMMA GEMM for odd/small shapes (N strips of 16).
template <int TN>
__global__ __launch_bounds__(32) void gemm_wmma_packed(
    const v16h* __restrict__ Ap, const v16h* __restrict__ Bp,
    const float* __restrict__ bias, float* __restrict__ C, int ldc,
    int M, int N, int KT)
{
  const int lane = threadIdx.x & 31;
  const int mt = blockIdx.y;
  const int nt0 = blockIdx.x * TN;
  const v16h* ap = Ap + (size_t)mt * KT * 32 + lane;
  const v16h* bp[TN];
  v8f acc[TN];
#pragma unroll
  for (int t = 0; t < TN; ++t) {
    bp[t] = Bp + (size_t)(nt0 + t) * KT * 32 + lane;
    acc[t] = (v8f){};
  }
  for (int kt = 0; kt < KT; ++kt) {
    v16h a = ap[(size_t)kt * 32];
    if (kt + 1 < KT) {
      __builtin_prefetch(ap + (size_t)(kt + 1) * 32, 0, 1);
      __builtin_prefetch(bp[0] + (size_t)(kt + 1) * 32, 0, 1);
    }
#pragma unroll
    for (int t = 0; t < TN; ++t) {
      v16h b = bp[t][(size_t)kt * 32];
      acc[t] = __builtin_amdgcn_wmma_f32_16x16x32_f16(
          false, a, false, b, (short)0, acc[t], false, false);
    }
  }
  const int l16 = lane & 15;
  const int rbase = mt * 16 + ((lane >> 4) << 3);
#pragma unroll
  for (int t = 0; t < TN; ++t) {
    const int col = (nt0 + t) * 16 + l16;
    if (col < N) {
      const float bv = bias ? bias[col] : 0.0f;
#pragma unroll
      for (int r = 0; r < 8; ++r) {
        const int rr = rbase + r;
        if (rr < M) C[(size_t)rr * ldc + col] = acc[t][r] + bv;
      }
    }
  }
}

// ---------------------------------------------------------------------------
// Positional embedding: feat[i] = [p(3), sin(2^e*pi*p) (24), cos(...) (24)]
// ---------------------------------------------------------------------------
__global__ __launch_bounds__(256) void embed_kernel(
    const float* __restrict__ x, float* __restrict__ p,
    float* __restrict__ feat, int n)
{
  int i = blockIdx.x * blockDim.x + threadIdx.x;
  if (i >= n) return;
  float ax[3] = {x[3 * i], x[3 * i + 1], x[3 * i + 2]};
  p[3 * i] = ax[0]; p[3 * i + 1] = ax[1]; p[3 * i + 2] = ax[2];
  float* f = feat + (size_t)i * 51;
  f[0] = ax[0]; f[1] = ax[1]; f[2] = ax[2];
#pragma unroll
  for (int d = 0; d < 3; ++d) {
#pragma unroll
    for (int e = 0; e < 8; ++e) {
      float fr = (float)(1 << e) * 3.14159265358979323846f * ax[d];
      f[3 + d * 8 + e]  = sinf(fr);
      f[27 + d * 8 + e] = cosf(fr);
    }
  }
}

// ---------------------------------------------------------------------------
// Train-mode BatchNorm: per-channel mean/var over rows (one block per channel)
// ---------------------------------------------------------------------------
__global__ __launch_bounds__(256) void col_stats_kernel(
    const float* __restrict__ X, int rows, int ld, int cols,
    float* __restrict__ mean, float* __restrict__ var)
{
  int c = blockIdx.x;
  if (c >= cols) return;
  __shared__ float s1[256];
  __shared__ float s2[256];
  int tid = threadIdx.x;
  float s = 0.0f, ss = 0.0f;
  for (int r = tid; r < rows; r += 256) {
    float v = X[(size_t)r * ld + c];
    s += v; ss += v * v;
  }
  s1[tid] = s; s2[tid] = ss;
  __syncthreads();
  for (int o = 128; o > 0; o >>= 1) {
    if (tid < o) { s1[tid] += s1[tid + o]; s2[tid] += s2[tid + o]; }
    __syncthreads();
  }
  if (tid == 0) {
    float m = s1[0] / (float)rows;
    mean[c] = m;
    var[c] = s2[0] / (float)rows - m * m;
  }
}

__global__ __launch_bounds__(256) void bn_apply_kernel(
    const float* __restrict__ X, int ldx, float* __restrict__ Y, int ldy,
    int rows, int cols,
    const float* __restrict__ mean, const float* __restrict__ var,
    const float* __restrict__ g, const float* __restrict__ b,
    const float* __restrict__ res, int ldr, int do_relu)
{
  size_t total = (size_t)rows * cols;
  size_t t = (size_t)blockIdx.x * blockDim.x + threadIdx.x;
  if (t >= total) return;
  int r = (int)(t / cols), c = (int)(t % cols);
  float v = (X[(size_t)r * ldx + c] - mean[c]) * rsqrtf(var[c] + EPSV) * g[c] + b[c];
  if (res) v += res[(size_t)r * ldr + c];
  if (do_relu) v = fmaxf(v, 0.0f);
  Y[(size_t)r * ldy + c] = v;
}

// ---------------------------------------------------------------------------
// Farthest point sampling: single workgroup, LDS argmax tree per iteration.
// ---------------------------------------------------------------------------
__global__ __launch_bounds__(256) void fps_kernel(
    const float* __restrict__ p, int n, int m,
    int* __restrict__ out_idx, float* __restrict__ dists)
{
  __shared__ float sb[256];
  __shared__ int   si[256];
  __shared__ int   s_last;
  int tid = threadIdx.x;
  for (int i = tid; i < n; i += 256) dists[i] = 1e10f;
  if (tid == 0) { out_idx[0] = 0; s_last = 0; }
  __syncthreads();
  for (int it = 1; it < m; ++it) {
    int last = s_last;
    float lx = p[3 * last], ly = p[3 * last + 1], lz = p[3 * last + 2];
    float best = -1.0f; int bidx = 0x7fffffff;
    for (int i = tid; i < n; i += 256) {
      float dx = p[3 * i] - lx, dy = p[3 * i + 1] - ly, dz = p[3 * i + 2] - lz;
      float d = dx * dx + dy * dy + dz * dz;
      float dm = fminf(dists[i], d);
      dists[i] = dm;
      if (dm > best) { best = dm; bidx = i; }
    }
    sb[tid] = best; si[tid] = bidx;
    __syncthreads();
    for (int o = 128; o > 0; o >>= 1) {
      if (tid < o) {
        if (sb[tid + o] > sb[tid] ||
            (sb[tid + o] == sb[tid] && si[tid + o] < si[tid])) {
          sb[tid] = sb[tid + o]; si[tid] = si[tid + o];
        }
      }
      __syncthreads();
    }
    if (tid == 0) { out_idx[it] = si[0]; s_last = si[0]; }
    __syncthreads();
  }
}

// ---------------------------------------------------------------------------
// Brute-force kNN: one query per thread; register-resident sorted top-K.
// ---------------------------------------------------------------------------
template <int K>
__global__ __launch_bounds__(128) void knn_kernel(
    const float* __restrict__ q, int nq,
    const float* __restrict__ p, int np,
    int* __restrict__ out)
{
  int t = blockIdx.x * blockDim.x + threadIdx.x;
  if (t >= nq) return;
  float qx = q[3 * t], qy = q[3 * t + 1], qz = q[3 * t + 2];
  float bd[K]; int bi[K];
#pragma unroll
  for (int i = 0; i < K; ++i) { bd[i] = 3.4e38f; bi[i] = 0; }
  for (int j = 0; j < np; ++j) {
    float dx = p[3 * j] - qx, dy = p[3 * j + 1] - qy, dz = p[3 * j + 2] - qz;
    float d = dx * dx + dy * dy + dz * dz;
    if (d < bd[K - 1]) {
      bd[K - 1] = d; bi[K - 1] = j;
#pragma unroll
      for (int s = K - 1; s > 0; --s) {
        if (bd[s] < bd[s - 1]) {
          float td = bd[s]; bd[s] = bd[s - 1]; bd[s - 1] = td;
          int ti = bi[s]; bi[s] = bi[s - 1]; bi[s - 1] = ti;
        }
      }
    }
  }
#pragma unroll
  for (int i = 0; i < K; ++i) out[(size_t)t * K + i] = bi[i];
}

// ---------------------------------------------------------------------------
// Misc data-movement kernels
// ---------------------------------------------------------------------------
__global__ __launch_bounds__(256) void gather_rows_kernel(
    const float* __restrict__ in, const int* __restrict__ idx,
    float* __restrict__ out, int rows, int cols)
{
  int t = blockIdx.x * blockDim.x + threadIdx.x;
  if (t >= rows * cols) return;
  int r = t / cols, c = t % cols;
  out[t] = in[(size_t)idx[r] * cols + c];
}

__global__ __launch_bounds__(256) void gather_td_kernel(
    const float* __restrict__ p, const float* __restrict__ newp,
    const float* __restrict__ x, int ldx, const int* __restrict__ idx,
    int m, int ns, int cin, float* __restrict__ out)
{
  int t = blockIdx.x * blockDim.x + threadIdx.x;
  if (t >= m * ns) return;
  int i = t / ns;
  int id = idx[t];
  float* o = out + (size_t)t * (3 + cin);
  o[0] = p[3 * id]     - newp[3 * i];
  o[1] = p[3 * id + 1] - newp[3 * i + 1];
  o[2] = p[3 * id + 2] - newp[3 * i + 2];
  for (int c = 0; c < cin; ++c) o[3 + c] = x[(size_t)id * ldx + c];
}

__global__ __launch_bounds__(256) void rowmax_kernel(
    const float* __restrict__ h, int m, int ns, int c, float* __restrict__ out)
{
  int t = blockIdx.x * blockDim.x + threadIdx.x;
  if (t >= m * c) return;
  int i = t / c, ch = t % c;
  float mx = -3.4e38f;
  for (int j = 0; j < ns; ++j)
    mx = fmaxf(mx, h[((size_t)i * ns + j) * c + ch]);
  out[t] = mx;
}

__global__ __launch_bounds__(256) void rel_kernel(
    const float* __restrict__ p, const int* __restrict__ idx,
    int n, int ns, float* __restrict__ rel)
{
  int t = blockIdx.x * blockDim.x + threadIdx.x;
  if (t >= n * ns) return;
  int i = t / ns;
  int id = idx[t];
  rel[3 * t]     = p[3 * id]     - p[3 * i];
  rel[3 * t + 1] = p[3 * id + 1] - p[3 * i + 1];
  rel[3 * t + 2] = p[3 * id + 2] - p[3 * i + 2];
}

__global__ __launch_bounds__(256) void w_assemble_kernel(
    const float* __restrict__ kf, const float* __restrict__ q,
    const float* __restrict__ pr, const int* __restrict__ idx,
    int n, int ns, int c, float* __restrict__ wb)
{
  size_t total = (size_t)n * ns * c;
  size_t t = (size_t)blockIdx.x * blockDim.x + threadIdx.x;
  if (t >= total) return;
  size_t row = t / c;
  int ch = (int)(t % c);
  int i = (int)(row / ns);
  wb[t] = kf[(size_t)idx[row] * c + ch] - q[(size_t)i * c + ch] + pr[t];
}

__global__ __launch_bounds__(256) void softmax_ns_kernel(
    float* __restrict__ w, int n, int ns, int cs)
{
  int t = blockIdx.x * blockDim.x + threadIdx.x;
  if (t >= n * cs) return;
  int i = t / cs, ch = t % cs;
  float* base = w + (size_t)i * ns * cs + ch;
  float mx = -3.4e38f;
  for (int j = 0; j < ns; ++j) mx = fmaxf(mx, base[(size_t)j * cs]);
  float s = 0.0f;
  for (int j = 0; j < ns; ++j) {
    float e = expf(base[(size_t)j * cs] - mx);
    base[(size_t)j * cs] = e;
    s += e;
  }
  float inv = 1.0f / s;
  for (int j = 0; j < ns; ++j) base[(size_t)j * cs] *= inv;
}

// out[i, cf] = sum_j (v[idx[i,j], cf] + pr[i,j,cf]) * w[i,j,cf % (c/share)]
__global__ __launch_bounds__(256) void attn_out_kernel(
    const float* __restrict__ vfeat, const float* __restrict__ pr,
    const float* __restrict__ w, const int* __restrict__ idx,
    int n, int ns, int c, int cs, float* __restrict__ out)
{
  size_t t = (size_t)blockIdx.x * blockDim.x + threadIdx.x;
  if (t >= (size_t)n * c) return;
  int i = (int)(t / c), cf = (int)(t % c);
  int ch = cf % cs;
  float s = 0.0f;
  for (int j = 0; j < ns; ++j) {
    size_t row = (size_t)i * ns + j;
    s += (vfeat[(size_t)idx[row] * c + cf] + pr[row * c + cf]) * w[row * cs + ch];
  }
  out[t] = s;
}

__global__ __launch_bounds__(256) void copy2_kernel(
    const float* __restrict__ a, int na, const float* __restrict__ b, int nb,
    float* __restrict__ out, int out_total)
{
  int t = blockIdx.x * blockDim.x + threadIdx.x;
  if (t >= out_total) return;
  if (t < na) out[t] = a[t];
  else if (t < na + nb) out[t] = b[t - na];
  else out[t] = 0.0f;
}

// ---------------------------------------------------------------------------
// Host orchestration
// ---------------------------------------------------------------------------
struct Bump {
  char* base;
  size_t off;
  void* get(size_t bytes) {
    void* p = base + off;
    off += (bytes + 255) & ~(size_t)255;
    return p;
  }
  float* f(size_t n) { return (float*)get(n * 4); }
  int*   i(size_t n) { return (int*)get(n * 4); }
};

static v16h* pack_a(hipStream_t s, Bump& ws, const float* A, int lda, int M, int K) {
  int MT = cdiv(M, 16), KT = cdiv(K, 32);
  v16h* Ap = (v16h*)ws.get((size_t)MT * KT * 1024);
  int total = MT * KT * 32;
  pack_a_kernel<<<cdiv(total, 256), 256, 0, s>>>(A, lda, M, K, KT, Ap, total);
  return Ap;
}

static v16h* pack_b(hipStream_t s, Bump& ws, const float* W, int ldw, int N, int K) {
  int NT = cdiv(N, 16), KT = cdiv(K, 32);
  v16h* Bp = (v16h*)ws.get((size_t)NT * KT * 1024);
  int total = NT * KT * 32;
  pack_b_kernel<<<cdiv(total, 256), 256, 0, s>>>(W, ldw, N, K, KT, Bp, total);
  return Bp;
}

static void gemm_run(hipStream_t s, const v16h* Ap, const v16h* Bp,
                     const float* bias, float* C, int ldc, int M, int N, int K) {
  int MT = cdiv(M, 16), NT = cdiv(N, 16), KT = cdiv(K, 32);
  if (N % 64 == 0) {
    dim3 g(NT / 4, cdiv(MT, 8));
    gemm_wmma_lds<4><<<g, 256, 0, s>>>(Ap, Bp, bias, C, ldc, M, N, KT, MT);
  } else {
    dim3 g(NT, MT);
    gemm_wmma_packed<1><<<g, 32, 0, s>>>(Ap, Bp, bias, C, ldc, M, N, KT);
  }
}

static void gemm_full(hipStream_t s, Bump& ws, const float* A, int lda,
                      const float* W, int ldw, const float* bias,
                      float* C, int ldc, int M, int N, int K) {
  v16h* Ap = pack_a(s, ws, A, lda, M, K);
  v16h* Bp = pack_b(s, ws, W, ldw, N, K);
  gemm_run(s, Ap, Bp, bias, C, ldc, M, N, K);
}

static void launch_stats(hipStream_t s, const float* X, int rows, int ld,
                         int cols, float* mean, float* var) {
  col_stats_kernel<<<cols, 256, 0, s>>>(X, rows, ld, cols, mean, var);
}

static void launch_bn(hipStream_t s, const float* X, int ldx, float* Y, int ldy,
                      int rows, int cols, const float* mean, const float* var,
                      const float* g, const float* b,
                      const float* res, int ldr, int relu) {
  long long total = (long long)rows * cols;
  bn_apply_kernel<<<cdiv(total, 256), 256, 0, s>>>(X, ldx, Y, ldy, rows, cols,
                                                   mean, var, g, b, res, ldr, relu);
}

struct Blk {
  const float *lin1_w, *bn1_g, *bn1_b;
  const float *q_w, *q_b, *k_w, *k_b, *v_w, *v_b;
  const float *p1_w, *p1_b, *pbn_g, *pbn_b, *p2_w, *p2_b;
  const float *wbn1_g, *wbn1_b, *w1_w, *w1_b, *wbn2_g, *wbn2_b, *w2_w, *w2_b;
  const float *bn2_g, *bn2_b, *lin3_w, *bn3_g, *bn3_b;
};

extern "C" void kernel_launch(void* const* d_in, const int* in_sizes, int n_in,
                              void* d_out, int out_size, void* d_ws, size_t ws_size,
                              hipStream_t stream)
{
  (void)in_sizes; (void)n_in; (void)ws_size;
  const int N0 = 8192;
  const int PLANES[3]  = {128, 256, 1024};
  const int STRIDEA[3] = {1, 4, 4};
  const int NSAMP[3]   = {8, 16, 16};
  const int NBLK[3]    = {1, 2, 3};  // BLOCKS[i]-1 residual blocks per stage

  // ---- unpack parameters (setup_inputs() dict insertion order) ----
  int gi = 0;
  const float* x_in = (const float*)d_in[gi++];
  const float* td_w[3]; const float* td_g[3]; const float* td_b[3];
  Blk blks[3][3];
  for (int s = 0; s < 3; ++s) {
    td_w[s] = (const float*)d_in[gi++];
    td_g[s] = (const float*)d_in[gi++];
    td_b[s] = (const float*)d_in[gi++];
    for (int j = 0; j < NBLK[s]; ++j) {
      Blk& B = blks[s][j];
      B.lin1_w = (const float*)d_in[gi++];
      B.bn1_g  = (const float*)d_in[gi++]; B.bn1_b  = (const float*)d_in[gi++];
      B.q_w = (const float*)d_in[gi++];    B.q_b = (const float*)d_in[gi++];
      B.k_w = (const float*)d_in[gi++];    B.k_b = (const float*)d_in[gi++];
      B.v_w = (const float*)d_in[gi++];    B.v_b = (const float*)d_in[gi++];
      B.p1_w = (const float*)d_in[gi++];   B.p1_b = (const float*)d_in[gi++];
      B.pbn_g = (const float*)d_in[gi++];  B.pbn_b = (const float*)d_in[gi++];
      B.p2_w = (const float*)d_in[gi++];   B.p2_b = (const float*)d_in[gi++];
      B.wbn1_g = (const float*)d_in[gi++]; B.wbn1_b = (const float*)d_in[gi++];
      B.w1_w = (const float*)d_in[gi++];   B.w1_b = (const float*)d_in[gi++];
      B.wbn2_g = (const float*)d_in[gi++]; B.wbn2_b = (const float*)d_in[gi++];
      B.w2_w = (const float*)d_in[gi++];   B.w2_b = (const float*)d_in[gi++];
      B.bn2_g = (const float*)d_in[gi++];  B.bn2_b = (const float*)d_in[gi++];
      B.lin3_w = (const float*)d_in[gi++];
      B.bn3_g = (const float*)d_in[gi++];  B.bn3_b = (const float*)d_in[gi++];
    }
  }

  Bump ws{(char*)d_ws, 0};

  // persistent ping-pong buffers (sized for the largest stage)
  float* pA = ws.f((size_t)N0 * 3);
  float* pB = ws.f((size_t)N0 * 3);
  float* fA = ws.f((size_t)N0 * 128);
  float* fB = ws.f((size_t)N0 * 128);
  size_t tempBase = ws.off;

  float* p_cur = pA;
  float* f_cur = fA;
  int n_cur = N0, c_cur = 51;

  embed_kernel<<<cdiv(N0, 256), 256, 0, stream>>>(x_in, p_cur, f_cur, N0);

  for (int s = 0; s < 3; ++s) {
    const int c = PLANES[s], st = STRIDEA[s], ns = NSAMP[s];
    ws.off = tempBase;
    float* mean = ws.f(1024);
    float* var  = ws.f(1024);
    size_t postStats = ws.off;

    if (st == 1) {
      // transition-down, stride 1: feat = relu(bn(feat @ W))
      float* f_next = (f_cur == fA) ? fB : fA;
      float* h = ws.f((size_t)n_cur * c);
      gemm_full(stream, ws, f_cur, c_cur, td_w[s], c, nullptr, h, c, n_cur, c, c_cur);
      launch_stats(stream, h, n_cur, c, c, mean, var);
      launch_bn(stream, h, c, f_next, c, n_cur, c, mean, var, td_g[s], td_b[s],
                nullptr, 0, 1);
      f_cur = f_next; c_cur = c;
    } else {
      // FPS -> kNN -> gather(rel xyz + feat) -> GEMM -> BN+ReLU -> max over ns
      const int m = n_cur / st;
      float* dists = ws.f((size_t)n_cur);
      int* fidx = ws.i((size_t)m);
      fps_kernel<<<1, 256, 0, stream>>>(p_cur, n_cur, m, fidx, dists);
      float* p_next = (p_cur == pA) ? pB : pA;
      gather_rows_kernel<<<cdiv((long long)m * 3, 256), 256, 0, stream>>>(
          p_cur, fidx, p_next, m, 3);
      int* tdidx = ws.i((size_t)m * ns);
      knn_kernel<16><<<cdiv(m, 128), 128, 0, stream>>>(p_next, m, p_cur, n_cur, tdidx);
      const int kin = 3 + c_cur;
      float* tdin = ws.f((size_t)m * ns * kin);
      gather_td_kernel<<<cdiv((long long)m * ns, 256), 256, 0, stream>>>(
          p_cur, p_next, f_cur, c_cur, tdidx, m, ns, c_cur, tdin);
      float* h = ws.f((size_t)m * ns * c);
      gemm_full(stream, ws, tdin, kin, td_w[s], c, nullptr, h, c, m * ns, c, kin);
      launch_stats(stream, h, m * ns, c, c, mean, var);
      launch_bn(stream, h, c, h, c, m * ns, c, mean, var, td_g[s], td_b[s],
                nullptr, 0, 1);
      float* f_next = (f_cur == fA) ? fB : fA;
      rowmax_kernel<<<cdiv((long long)m * c, 256), 256, 0, stream>>>(h, m, ns, c, f_next);
      p_cur = p_next; f_cur = f_next; n_cur = m; c_cur = c;
    }

    // self-kNN reused by all blocks in this stage
    ws.off = postStats;
    int* sidx = ws.i((size_t)n_cur * ns);
    if (ns == 8)
      knn_kernel<8><<<cdiv(n_cur, 128), 128, 0, stream>>>(p_cur, n_cur, p_cur, n_cur, sidx);
    else
      knn_kernel<16><<<cdiv(n_cur, 128), 128, 0, stream>>>(p_cur, n_cur, p_cur, n_cur, sidx);
    size_t blockBase = ws.off;

    for (int j = 0; j < NBLK[s]; ++j) {
      ws.off = blockBase;
      const Blk& B = blks[s][j];
      const int n = n_cur, C = c, CS = c / 8;
      const size_t nc = (size_t)n * C;
      const size_t nns = (size_t)n * ns;

      // h1 = relu(bn1(x @ lin1))
      float* h1 = ws.f(nc);
      gemm_full(stream, ws, f_cur, C, B.lin1_w, C, nullptr, h1, C, n, C, C);
      launch_stats(stream, h1, n, C, C, mean, var);
      launch_bn(stream, h1, C, h1, C, n, C, mean, var, B.bn1_g, B.bn1_b, nullptr, 0, 1);

      // q, k, v projections: pack A (h1) once, three packed GEMMs
      float* qf = ws.f(nc);
      float* kf = ws.f(nc);
      float* vf = ws.f(nc);
      {
        v16h* Ah = pack_a(stream, ws, h1, C, n, C);
        v16h* Bq = pack_b(stream, ws, B.q_w, C, C, C);
        gemm_run(stream, Ah, Bq, B.q_b, qf, C, n, C, C);
        v16h* Bk = pack_b(stream, ws, B.k_w, C, C, C);
        gemm_run(stream, Ah, Bk, B.k_b, kf, C, n, C, C);
        v16h* Bv = pack_b(stream, ws, B.v_w, C, C, C);
        gemm_run(stream, Ah, Bv, B.v_b, vf, C, n, C, C);
      }

      // positional encoding: pr = (relu(bn(rel @ p1)) @ p2)
      float* rel = ws.f(nns * 3);
      rel_kernel<<<cdiv((long long)nns, 256), 256, 0, stream>>>(p_cur, sidx, n, ns, rel);
      float* pr1 = ws.f(nns * 3);
      gemm_full(stream, ws, rel, 3, B.p1_w, 3, B.p1_b, pr1, 3, (int)nns, 3, 3);
      launch_stats(stream, pr1, (int)nns, 3, 3, mean, var);
      launch_bn(stream, pr1, 3, pr1, 3, (int)nns, 3, mean, var, B.pbn_g, B.pbn_b,
                nullptr, 0, 1);
      float* pr = ws.f(nns * C);
      gemm_full(stream, ws, pr1, 3, B.p2_w, C, B.p2_b, pr, C, (int)nns, C, 3);

      // attention weights: w = softmax_ns(w2(relu(bn(w1(relu(bn(k-q+pr)))))))
      float* wb = ws.f(nns * C);
      w_assemble_kernel<<<cdiv((long long)nns * C, 256), 256, 0, stream>>>(
          kf, qf, pr, sidx, n, ns, C, wb);
      launch_stats(stream, wb, (int)nns, C, C, mean, var);
      launch_bn(stream, wb, C, wb, C, (int)nns, C, mean, var, B.wbn1_g, B.wbn1_b,
                nullptr, 0, 1);
      float* w1o = ws.f(nns * CS);
      gemm_full(stream, ws, wb, C, B.w1_w, CS, B.w1_b, w1o, CS, (int)nns, CS, C);
      launch_stats(stream, w1o, (int)nns, CS, CS, mean, var);
      launch_bn(stream, w1o, CS, w1o, CS, (int)nns, CS, mean, var, B.wbn2_g, B.wbn2_b,
                nullptr, 0, 1);
      float* w2o = ws.f(nns * CS);
      gemm_full(stream, ws, w1o, CS, B.w2_w, CS, B.w2_b, w2o, CS, (int)nns, CS, CS);
      softmax_ns_kernel<<<cdiv((long long)n * CS, 256), 256, 0, stream>>>(w2o, n, ns, CS);

      // aggregate: y = sum_ns (v[idx]+pr) * w  (grouped over share)
      float* y = ws.f(nc);
      attn_out_kernel<<<cdiv((long long)nc, 256), 256, 0, stream>>>(
          vf, pr, w2o, sidx, n, ns, C, CS, y);
      launch_stats(stream, y, n, C, C, mean, var);
      launch_bn(stream, y, C, y, C, n, C, mean, var, B.bn2_g, B.bn2_b, nullptr, 0, 1);

      // out = relu(bn3(y @ lin3) + x)   (residual, in-place into f_cur)
      float* h3 = ws.f(nc);
      gemm_full(stream, ws, y, C, B.lin3_w, C, nullptr, h3, C, n, C, C);
      launch_stats(stream, h3, n, C, C, mean, var);
      launch_bn(stream, h3, C, f_cur, C, n, C, mean, var, B.bn3_g, B.bn3_b,
                f_cur, C, 1);
    }
  }

  // output: concat(p (512x3), feat (512x1024)) flat
  const int np = n_cur * 3;
  const int nf = n_cur * c_cur;
  copy2_kernel<<<cdiv(out_size, 256), 256, 0, stream>>>(
      p_cur, np, f_cur, nf, (float*)d_out, out_size);
}
